// FastSelfAttention_29575144801170
// MI455X (gfx1250) — compile-verified, matrix operand order checked by reading, not run on hardware
//
#include <hip/hip_runtime.h>
#include <stdint.h>
#include <math.h>

#define B_   8
#define S_   4096
#define H_   1024
#define NH_  16
#define HD_  64
#define SCALE_ 0.125f            // 1/sqrt(64)
#define BSH_ ((size_t)B_ * S_ * H_)

typedef __attribute__((ext_vector_type(16))) __bf16 v16bf;
typedef __attribute__((ext_vector_type(8)))  __bf16 v8bf;
typedef __attribute__((ext_vector_type(8)))  float  v8f;
typedef __attribute__((ext_vector_type(4)))  float  v4f;
typedef __attribute__((ext_vector_type(4)))  unsigned int u32x4;
typedef __attribute__((ext_vector_type(8)))  int    i32x8;
typedef __attribute__((ext_vector_type(4)))  int    i32x4;

static __device__ __forceinline__ v8f zero8() {
    v8f z;
#pragma unroll
    for (int i = 0; i < 8; ++i) z[i] = 0.f;
    return z;
}

// ---------------------------------------------------------------------------
// TDM: load a 64-row x 32-col bf16 tile (row stride 1024 elems in memory)
// into LDS with 16B padding after each 64B row  ->  LDS row stride = 40 bf16.
// ---------------------------------------------------------------------------
static __device__ __forceinline__ void tdm_load_btile(uint32_t lds_byte_off,
                                                      const __bf16* gptr) {
    uint64_t ga = (uint64_t)(uintptr_t)gptr;
    u32x4 g0;
    g0[0] = 1u;                                              // count=1
    g0[1] = lds_byte_off;                                    // lds_addr
    g0[2] = (uint32_t)ga;                                    // global_addr lo
    g0[3] = (uint32_t)((ga >> 32) & 0x01FFFFFFu) | (2u << 30); // ga hi | type=2
    i32x8 g1;
    g1[0] = (int)((1u << 16) |      // data_size = 2 bytes
                  (1u << 20) |      // pad_enable
                  (3u << 22) |      // pad_interval: 16 DWORDs (64B)
                  (3u << 25));      // pad_amount:   4 DWORDs (16B)
    g1[1] = (int)((uint32_t)(H_ & 0xFFFF) << 16);            // tensor_dim0 lo16
    g1[2] = (int)((uint32_t)64u << 16);                      // dim0 hi16=0 | tensor_dim1 lo16=64
    g1[3] = (int)((uint32_t)32u << 16);                      // dim1 hi16=0 | tile_dim0=32
    g1[4] = (int)64u;                                        // tile_dim1=64 | tile_dim2=0
    g1[5] = (int)H_;                                         // tensor_dim0_stride lo32
    g1[6] = 0;                                               // stride hi16 | dim1_stride lo16
    g1[7] = 0;
    i32x4 gz;
    gz[0] = 0; gz[1] = 0; gz[2] = 0; gz[3] = 0;
    i32x8 gz8;
#pragma unroll
    for (int i = 0; i < 8; ++i) gz8[i] = 0;
    __builtin_amdgcn_tensor_load_to_lds(g0, g1, gz, gz, gz8, 0);
}

// ---------------------------------------------------------------------------
// Weight pack: W [K=H][N=H] fp32 row-major  ->  WT [N][K] bf16 row-major
// ---------------------------------------------------------------------------
__global__ __launch_bounds__(256) void pack_w_kernel(const float* __restrict__ W,
                                                     __bf16* __restrict__ WT) {
    int idx = blockIdx.x * 256 + threadIdx.x;      // over H*H, n-major / k-fast
    int n = idx >> 10;
    int k = idx & (H_ - 1);
    WT[idx] = (__bf16)W[(size_t)k * H_ + n];
}

// ---------------------------------------------------------------------------
// Tiled bf16-WMMA GEMM, double-buffered LDS + 2-deep A-load pipeline;
// B tiles staged by the Tensor Data Mover.
//   MODE 0 (dual):   out1 = A@W1 + bias1 ; out2 = A@W2 + bias2   (A read once)
//   MODE 1 (single): out1 = (A * colScale[b,k]) @ W1 + bias1 + residual
// ---------------------------------------------------------------------------
template <int MODE>
__global__ __launch_bounds__(256) void gemm_kernel(
    const float* __restrict__ A, const float* __restrict__ colScale,
    const __bf16* __restrict__ B1T, const float* __restrict__ bias1,
    float* __restrict__ out1,
    const __bf16* __restrict__ B2T, const float* __restrict__ bias2,
    float* __restrict__ out2, const float* __restrict__ residual) {
    constexpr int BM = 128, BN = 64, BK = 32, LDA = BK + 8;  // pad -> stride 40
    constexpr int NT = H_ / BK;                              // 32 K-tiles
    constexpr int ASZ = BM * LDA, BSZ = BN * LDA;
    __shared__ __align__(16) __bf16 As[2 * ASZ];
    __shared__ __align__(16) __bf16 Bs1[2 * BSZ];
    __shared__ __align__(16) __bf16 Bs2[(MODE == 0 ? 2 * BSZ : 1)];

    const int tid  = threadIdx.x;
    const int lane = tid & 31;
    const int wave = tid >> 5;
    const int wm   = wave & 1;        // 0..1  -> 64-row half
    const int wn   = wave >> 1;       // 0..3  -> 16-col slice
    const int m0   = blockIdx.y * BM;
    const int n0   = blockIdx.x * BN;
    const int batch = m0 / S_;        // BM=128 divides S, block stays in-batch

    const uint32_t ldsB1 = (uint32_t)(uintptr_t)(void*)Bs1;
    const uint32_t ldsB2 = (uint32_t)(uintptr_t)(void*)Bs2;

    // per-thread A-staging coordinates (loop-invariant)
    int ar[4], ac[4];
    const float* agp[4];
#pragma unroll
    for (int j = 0; j < 4; ++j) {
        int q = tid + j * 256;            // 1024 float4s of a 128x32 tile
        ar[j] = q >> 3;
        ac[j] = (q & 7) * 4;
        agp[j] = A + (size_t)(m0 + ar[j]) * H_ + ac[j];
    }
    const float* scp = (MODE == 1) ? (colScale + (size_t)batch * H_) : nullptr;

    v8f acc1[4], acc2[MODE == 0 ? 4 : 1];
#pragma unroll
    for (int i = 0; i < 4; ++i) acc1[i] = zero8();
    if (MODE == 0) {
#pragma unroll
        for (int i = 0; i < (MODE == 0 ? 4 : 1); ++i) acc2[i] = zero8();
    }

    const int frow = lane & 15;           // M (A) / N (B) index inside a tile
    const int koff = (lane >> 4) * 8;     // lane<16: K {0..7,16..23}; else {8..15,24..31}

    v4f areg[4], sreg[4];

    auto issue_tdm = [&](int buf, int kt) {
        if (wave == 0) {
            tdm_load_btile(ldsB1 + (uint32_t)(buf * BSZ * 2),
                           B1T + (size_t)n0 * H_ + kt);
            if (MODE == 0)
                tdm_load_btile(ldsB2 + (uint32_t)(buf * BSZ * 2),
                               B2T + (size_t)n0 * H_ + kt);
        }
    };
    auto issue_loads = [&](int kt) {
#pragma unroll
        for (int j = 0; j < 4; ++j) {
            areg[j] = *(const v4f*)(agp[j] + kt);
            if (MODE == 1) sreg[j] = *(const v4f*)(scp + kt + ac[j]);
        }
    };
    auto store_tile = [&](int buf) {
#pragma unroll
        for (int j = 0; j < 4; ++j) {
            v4f v = areg[j];
            if (MODE == 1) {
                v[0] *= sreg[j][0]; v[1] *= sreg[j][1];
                v[2] *= sreg[j][2]; v[3] *= sreg[j][3];
            }
            __bf16* dst = As + buf * ASZ + ar[j] * LDA + ac[j];
            dst[0] = (__bf16)v[0]; dst[1] = (__bf16)v[1];
            dst[2] = (__bf16)v[2]; dst[3] = (__bf16)v[3];
        }
    };

    // ---- prologue: stage tile 0, prefetch tile 1 into registers ----
    issue_tdm(0, 0);
    issue_loads(0);
    store_tile(0);
    issue_loads(BK);                      // NT = 32 >= 2 always

    for (int it = 0; it < NT; ++it) {
        const int cur = it & 1, nxt = cur ^ 1;
        if (wave == 0) __builtin_amdgcn_s_wait_tensorcnt(0);
        __syncthreads();   // publishes TDM + ds stores of tile `cur`;
                           // orders prior compute before overwriting `nxt`
        if (it + 1 < NT)
            issue_tdm(nxt, (it + 1) * BK);    // DMA next B tile during compute

        // --- B fragments (N row = wn*16 + frow, K {koff..koff+7, 16+koff..16+koff+7}) ---
        const __bf16* bb1 = Bs1 + cur * BSZ + (wn * 16 + frow) * LDA;
        v16bf bf1;
        *(v8bf*)&bf1       = *(const v8bf*)(bb1 + koff);
        *((v8bf*)&bf1 + 1) = *(const v8bf*)(bb1 + 16 + koff);
        v16bf bf2;
        if (MODE == 0) {
            const __bf16* bb2 = Bs2 + cur * BSZ + (wn * 16 + frow) * LDA;
            *(v8bf*)&bf2       = *(const v8bf*)(bb2 + koff);
            *((v8bf*)&bf2 + 1) = *(const v8bf*)(bb2 + 16 + koff);
        }
        // --- A fragments + WMMA ---
#pragma unroll
        for (int mt = 0; mt < 4; ++mt) {
            const __bf16* ab = As + cur * ASZ + (wm * 64 + mt * 16 + frow) * LDA;
            v16bf af;
            *(v8bf*)&af       = *(const v8bf*)(ab + koff);
            *((v8bf*)&af + 1) = *(const v8bf*)(ab + 16 + koff);
            acc1[mt] = __builtin_amdgcn_wmma_f32_16x16x32_bf16(
                false, af, false, bf1, (short)0, acc1[mt], false, false);
            if (MODE == 0)
                acc2[mt] = __builtin_amdgcn_wmma_f32_16x16x32_bf16(
                    false, af, false, bf2, (short)0, acc2[mt], false, false);
        }
        // cvt+store tile `nxt` from registers loaded a full iteration ago,
        // then prefetch tile `it+2` into those registers
        if (it + 1 < NT) store_tile(nxt);
        if (it + 2 < NT) issue_loads((it + 2) * BK);
    }

    // --- epilogue: C/D layout -> lane<16: M=v, N=lane ; lane>=16: M=v+8 ---
    const int col = n0 + wn * 16 + frow;
    const float b1v = bias1[col];
    const float b2v = (MODE == 0) ? bias2[col] : 0.f;
#pragma unroll
    for (int mt = 0; mt < 4; ++mt) {
        int rbase = m0 + wm * 64 + mt * 16 + ((lane < 16) ? 0 : 8);
#pragma unroll
        for (int v = 0; v < 8; ++v) {
            size_t idx = (size_t)(rbase + v) * H_ + col;
            float x = acc1[mt][v] + b1v;
            if (MODE == 1) x += residual[idx];
            out1[idx] = x;
            if (MODE == 0) out2[idx] = acc2[mt][v] + b2v;
        }
    }
}

// ---------------------------------------------------------------------------
// scores[b,nh,s] = (sum_h Min[b,s,h]*(colScale?colScale[b,h]:1)*Wa[h,nh]
//                  + ba[nh]) * SCALE + mask[b,s]
// One wave per (b,s) row; Wa (H x 16) staged in LDS.
// ---------------------------------------------------------------------------
__global__ __launch_bounds__(256) void score_kernel(
    const float* __restrict__ Min, const float* __restrict__ colScale,
    const float* __restrict__ Wa, const float* __restrict__ ba,
    const float* __restrict__ mask, float* __restrict__ scores) {
    __shared__ float sW[H_ * NH_];                 // 64 KB
    const int tid = threadIdx.x;
    for (int i = tid; i < H_ * NH_; i += 256) sW[i] = Wa[i];
    __syncthreads();

    const int lane = tid & 31, wave = tid >> 5;
    const int row = blockIdx.x * 8 + wave;         // b*S + s
    const int b = row / S_, s = row & (S_ - 1);

    float acc[NH_];
#pragma unroll
    for (int j = 0; j < NH_; ++j) acc[j] = 0.f;

    for (int h = lane; h < H_; h += 32) {
        float x = Min[(size_t)row * H_ + h];
        if (colScale) x *= colScale[(size_t)b * H_ + h];
        const float* wr = sW + h * NH_;
#pragma unroll
        for (int j = 0; j < NH_; ++j) acc[j] += x * wr[j];
    }
#pragma unroll
    for (int j = 0; j < NH_; ++j)
        for (int off = 16; off > 0; off >>= 1)
            acc[j] += __shfl_down(acc[j], off, 32);

    if (lane == 0) {
        float mv = mask[(size_t)b * S_ + s];
#pragma unroll
        for (int j = 0; j < NH_; ++j)
            scores[((size_t)b * NH_ + j) * S_ + s] = (acc[j] + ba[j]) * SCALE_ + mv;
    }
}

// ---------------------------------------------------------------------------
// In-place softmax over S for each of B*NH rows (one block per row).
// ---------------------------------------------------------------------------
__global__ __launch_bounds__(256) void softmax_kernel(float* __restrict__ scores) {
    __shared__ float red[256];
    float* rowp = scores + (size_t)blockIdx.x * S_;
    const int tid = threadIdx.x;

    float m = -INFINITY;
    for (int i = tid; i < S_; i += 256) m = fmaxf(m, rowp[i]);
    red[tid] = m; __syncthreads();
    for (int off = 128; off > 0; off >>= 1) {
        if (tid < off) red[tid] = fmaxf(red[tid], red[tid + off]);
        __syncthreads();
    }
    m = red[0]; __syncthreads();

    float sum = 0.f;
    for (int i = tid; i < S_; i += 256) {
        float e = __expf(rowp[i] - m);
        rowp[i] = e;
        sum += e;
    }
    red[tid] = sum; __syncthreads();
    for (int off = 128; off > 0; off >>= 1) {
        if (tid < off) red[tid] += red[tid + off];
        __syncthreads();
    }
    float inv = 1.f / red[0];
    for (int i = tid; i < S_; i += 256) rowp[i] *= inv;
}

// ---------------------------------------------------------------------------
// outp[b*H + nh*64 + d] = (sum_s w[b,nh,s] * Min[b,s,nh*64+d]) * finalScale?
// One block per (b,nh); 256 threads = 64 d-lanes x 4 s-groups.
// ---------------------------------------------------------------------------
__global__ __launch_bounds__(256) void pool_kernel(
    const float* __restrict__ w, const float* __restrict__ Min,
    const float* __restrict__ finalScale, float* __restrict__ outp) {
    const int bn = blockIdx.x;            // b*NH + nh
    const int b = bn / NH_, nh = bn % NH_;
    const int tid = threadIdx.x;
    const int d = tid & 63, sg = tid >> 6;
    const float* wrow = w + (size_t)bn * S_;

    float acc = 0.f;
    for (int s = sg; s < S_; s += 4)
        acc += wrow[s] * Min[((size_t)b * S_ + s) * H_ + nh * HD_ + d];

    __shared__ float red[256];
    red[tid] = acc; __syncthreads();
    if (sg == 0) {
        float t = red[d] + red[64 + d] + red[128 + d] + red[192 + d];
        int hidx = nh * HD_ + d;
        if (finalScale) t *= finalScale[(size_t)b * H_ + hidx];
        outp[(size_t)b * H_ + hidx] = t;
    }
}

// ---------------------------------------------------------------------------
extern "C" void kernel_launch(void* const* d_in, const int* in_sizes, int n_in,
                              void* d_out, int out_size, void* d_ws, size_t ws_size,
                              hipStream_t stream) {
    (void)in_sizes; (void)n_in; (void)out_size; (void)ws_size;
    const float* X    = (const float*)d_in[0];
    const float* mask = (const float*)d_in[1];
    const float* Wq   = (const float*)d_in[2];
    const float* bq   = (const float*)d_in[3];
    const float* Wqa  = (const float*)d_in[4];
    const float* bqa  = (const float*)d_in[5];
    const float* Wk   = (const float*)d_in[6];
    const float* bk   = (const float*)d_in[7];
    const float* Wka  = (const float*)d_in[8];
    const float* bka  = (const float*)d_in[9];
    const float* Wt   = (const float*)d_in[10];
    const float* bt   = (const float*)d_in[11];
    float* out = (float*)d_out;

    // workspace carve-up
    char* ws = (char*)d_ws;
    float* mq = (float*)ws;                 ws += BSH_ * sizeof(float);
    float* mk = (float*)ws;                 ws += BSH_ * sizeof(float);
    float* scores = (float*)ws;             ws += (size_t)B_ * NH_ * S_ * sizeof(float);
    float* pq = (float*)ws;                 ws += (size_t)B_ * H_ * sizeof(float);
    float* pk = (float*)ws;                 ws += (size_t)B_ * H_ * sizeof(float);
    __bf16* WqT = (__bf16*)ws;              ws += (size_t)H_ * H_ * sizeof(__bf16);
    __bf16* WkT = (__bf16*)ws;              ws += (size_t)H_ * H_ * sizeof(__bf16);
    __bf16* WtT = (__bf16*)ws;              ws += (size_t)H_ * H_ * sizeof(__bf16);

    const dim3 blk(256);
    const dim3 packGrid((H_ * H_) / 256);
    pack_w_kernel<<<packGrid, blk, 0, stream>>>(Wq, WqT);
    pack_w_kernel<<<packGrid, blk, 0, stream>>>(Wk, WkT);
    pack_w_kernel<<<packGrid, blk, 0, stream>>>(Wt, WtT);

    const dim3 gemmGrid(H_ / 64, (B_ * S_) / 128);
    // mq = X@Wq + bq ; mk = X@Wk + bk   (X read once)
    gemm_kernel<0><<<gemmGrid, blk, 0, stream>>>(X, nullptr, WqT, bq, mq,
                                                 WkT, bk, mk, nullptr);

    const dim3 scoreGrid((B_ * S_) / 8);
    const dim3 rowGrid(B_ * NH_);
    // query pooling
    score_kernel<<<scoreGrid, blk, 0, stream>>>(mq, nullptr, Wqa, bqa, mask, scores);
    softmax_kernel<<<rowGrid, blk, 0, stream>>>(scores);
    pool_kernel<<<rowGrid, blk, 0, stream>>>(scores, mq, nullptr, pq);
    // key pooling (mqk = mk * pq folded in as per-column scale / final scale)
    score_kernel<<<scoreGrid, blk, 0, stream>>>(mk, pq, Wka, bka, mask, scores);
    softmax_kernel<<<rowGrid, blk, 0, stream>>>(scores);
    pool_kernel<<<rowGrid, blk, 0, stream>>>(scores, mk, pq, pk);

    // out = (mq * pk) @ Wt + bt + mq
    gemm_kernel<1><<<gemmGrid, blk, 0, stream>>>(mq, pk, WtT, bt, out,
                                                 nullptr, nullptr, nullptr, mq);
}
